// HMM_80410377716208
// MI455X (gfx1250) — matrix operand order repaired and checked by reading, not compiled, not run on hardware
//
#include <hip/hip_runtime.h>
#include <hip/hip_bf16.h>
#include <math.h>

// ---------------------------------------------------------------------------
// HMM ELBO (rec_loss, kl_loss) for B=128, T=512, D=512, Z=64.
//
// Structure: q(z) is shared across all (b,t), so the KL terms are two scalars
// and the reconstruction term is a masked GEMV:
//     rec[b,t] = -(S + sum_d x[b,t,d] * c[d]),   c = log(e+eps)-log(1-e+eps)
// Memory-bound: 134 MB of x at 23.3 TB/s ~ 5.8us (less with length skipping).
//
// Kernel 1 (1 block): softmaxes, kl0/klt, c[512], S, denom -> workspace.
// Kernel 2 (4096 waves): per-wave 16-row tile, D-long dots via
//     v_wmma_f32_16x16x32_f16, rows masked by x_lens (zeroed A rows,
//     whole-tile skip when fully past the length). Partials -> workspace.
// Kernel 3 (1 block): deterministic tree reduce + final scalars -> d_out.
// ---------------------------------------------------------------------------

typedef __attribute__((ext_vector_type(16))) _Float16 v16h;
typedef __attribute__((ext_vector_type(8)))  float    v8f;

#define HMM_B 128
#define HMM_T 512
#define HMM_D 512
#define HMM_Z 64
#define EPSF  1e-10f

#define N_TILES ((HMM_B * HMM_T) / 16)   // 4096 row-tiles of 16 rows
#define WS_C      0                       // ws[0..511]   : c[d]
#define WS_S      512                     // ws[512]      : S = sum log(1-e+eps)
#define WS_KL0    513
#define WS_KLT    514
#define WS_DENOM  515
#define WS_PART   1024                    // ws[1024..1024+4095] : wave partials

// ------------------------- Kernel 1: tiny precompute ------------------------
__global__ __launch_bounds__(512)
void hmm_prep(const int* __restrict__ x_lens,
              const float* __restrict__ trans,    // [Z,Z]
              const float* __restrict__ emis,     // [Z,D]
              const float* __restrict__ init,     // [Z]
              const float* __restrict__ qlog,     // [1,Z]
              float* __restrict__ ws)
{
    __shared__ float sq[HMM_Z];      // q probs
    __shared__ float smx[HMM_Z];     // per-row max of transition
    __shared__ float srs[HMM_Z];     // per-row 1/sumexp
    __shared__ float red[512];
    const int tid = threadIdx.x;

    // ---- deterministic block tree reductions over 512 slots ----
    auto reduce_add = [&](float v) -> float {
        red[tid] = v; __syncthreads();
        for (int s = 256; s > 0; s >>= 1) {
            if (tid < s) red[tid] += red[tid + s];
            __syncthreads();
        }
        float r = red[0]; __syncthreads(); return r;
    };
    auto reduce_max = [&](float v) -> float {
        red[tid] = v; __syncthreads();
        for (int s = 256; s > 0; s >>= 1) {
            if (tid < s) red[tid] = fmaxf(red[tid], red[tid + s]);
            __syncthreads();
        }
        float r = red[0]; __syncthreads(); return r;
    };

    // ---- q softmax ----
    float qv = (tid < HMM_Z) ? qlog[tid] : -3.0e38f;
    float qmx = reduce_max(qv);
    float qe = (tid < HMM_Z) ? __expf(qv - qmx) : 0.f;
    float qsum = reduce_add(qe);
    if (tid < HMM_Z) sq[tid] = qe / qsum;
    __syncthreads();

    // ---- kl0 vs softmax(initial_logits) ----
    float iv = (tid < HMM_Z) ? init[tid] : -3.0e38f;
    float imx = reduce_max(iv);
    float ie = (tid < HMM_Z) ? __expf(iv - imx) : 0.f;
    float isum = reduce_add(ie);
    float kl0t = 0.f;
    if (tid < HMM_Z) {
        float p0 = ie / isum;
        float q = sq[tid];
        kl0t = q * (__logf(q + EPSF) - __logf(p0 + EPSF));
    }
    float kl0 = reduce_add(kl0t);

    // ---- row softmax stats of transition ----
    if (tid < HMM_Z) {
        const float* row = trans + tid * HMM_Z;
        float mx = row[0];
        for (int j = 1; j < HMM_Z; ++j) mx = fmaxf(mx, row[j]);
        float s = 0.f;
        for (int j = 0; j < HMM_Z; ++j) s += __expf(row[j] - mx);
        smx[tid] = mx;
        srs[tid] = 1.f / s;
    }
    __syncthreads();

    // ---- p_next = q @ softmax(A);  klt vs normalize(p_next + eps) ----
    float pn = 0.f;
    if (tid < HMM_Z) {
        for (int z = 0; z < HMM_Z; ++z)
            pn += sq[z] * __expf(trans[z * HMM_Z + tid] - smx[z]) * srs[z];
        pn += EPSF;
    }
    float pnsum = reduce_add(pn);
    float kltt = 0.f;
    if (tid < HMM_Z) {
        float pnp = pn / pnsum;
        float q = sq[tid];
        kltt = q * (__logf(q + EPSF) - __logf(pnp + EPSF));
    }
    float klt = reduce_add(kltt);

    // ---- emission: e[d] = sum_z q[z] sigmoid(E[z,d]); c[d]; S ----
    {
        const int d = tid;               // 512 threads == D
        float e = 0.f;
        for (int z = 0; z < HMM_Z; ++z)
            e += sq[z] / (1.f + __expf(-emis[z * HMM_D + d]));
        float le  = __logf(e + EPSF);
        float l1e = __logf(1.f - e + EPSF);
        ws[WS_C + d] = le - l1e;
        float S = reduce_add(l1e);
        if (tid == 0) ws[WS_S] = S;
    }

    // ---- denom = sum x_lens ----
    float lv = (tid < HMM_B) ? (float)x_lens[tid] : 0.f;
    float denom = reduce_add(lv);

    if (tid == 0) {
        ws[WS_KL0]   = kl0;
        ws[WS_KLT]   = klt;
        ws[WS_DENOM] = denom;
    }
}

// --------------- Kernel 2: masked GEMV x @ c via WMMA f16 tiles -------------
// One wave per 16-row tile. A = 16x32 f32->f16 tile of x (K = 32 columns of D),
// B = c-segment replicated into all 16 columns. Accumulate 16 K-segments in
// v8f C. Dense 16-bit A 16x32 layout (ISA 7.12.2): lane<16 holds row M=lane,
// halves = K{kb+0..7, kb+16..23}; lane>=16 holds row M=lane-16, halves =
// K{kb+8..15, kb+24..31}. B (same K order as A): lane<16
// holds K{kb+0..7, kb+16..23}, lane>=16 holds K{kb+8..15, kb+24..31}; replication across N makes
// the result independent of the N->lane mapping.
__global__ __launch_bounds__(256)
void hmm_dot(const float* __restrict__ x,
             const int*   __restrict__ x_lens,
             const float* __restrict__ ws,
             float*       __restrict__ partials)
{
    const int lane   = threadIdx.x & 31;
    const int waveId = blockIdx.x * (blockDim.x >> 5) + (threadIdx.x >> 5);
    const int row0   = waveId * 16;                 // tile never straddles b
    const int b      = row0 >> 9;                   // /T
    const int t0     = row0 & (HMM_T - 1);
    const int len    = x_lens[b];                   // wave-uniform

    float tile_sum = 0.f;
    if (t0 < len) {                                 // uniform branch: EXEC full
        const int  mrow   = row0 + (lane & 15);
        const bool valid  = (mrow & (HMM_T - 1)) < len;
        const int  khalfA = (lane >= 16) ? 8  : 0;
        const int  khalfB = (lane >= 16) ? 8  : 0;
        const float* px = x + (size_t)mrow * HMM_D + khalfA;
        const float* pc = ws + WS_C + khalfB;

        v8f acc = {};
        #pragma unroll
        for (int seg = 0; seg < HMM_D / 32; ++seg) {
            const int kb = seg * 32;

            v16h a = {};
            if (valid) {                            // skip reads of dead rows
                float4 f0 = *(const float4*)(px + kb);
                float4 f1 = *(const float4*)(px + kb + 4);
                float4 f2 = *(const float4*)(px + kb + 16);
                float4 f3 = *(const float4*)(px + kb + 20);
                a[0]=(_Float16)f0.x;  a[1]=(_Float16)f0.y;
                a[2]=(_Float16)f0.z;  a[3]=(_Float16)f0.w;
                a[4]=(_Float16)f1.x;  a[5]=(_Float16)f1.y;
                a[6]=(_Float16)f1.z;  a[7]=(_Float16)f1.w;
                a[8]=(_Float16)f2.x;  a[9]=(_Float16)f2.y;
                a[10]=(_Float16)f2.z; a[11]=(_Float16)f2.w;
                a[12]=(_Float16)f3.x; a[13]=(_Float16)f3.y;
                a[14]=(_Float16)f3.z; a[15]=(_Float16)f3.w;
            }

            float4 c0 = *(const float4*)(pc + kb);
            float4 c1 = *(const float4*)(pc + kb + 4);
            float4 c2 = *(const float4*)(pc + kb + 16);
            float4 c3 = *(const float4*)(pc + kb + 20);
            v16h bm;
            bm[0]=(_Float16)c0.x;  bm[1]=(_Float16)c0.y;
            bm[2]=(_Float16)c0.z;  bm[3]=(_Float16)c0.w;
            bm[4]=(_Float16)c1.x;  bm[5]=(_Float16)c1.y;
            bm[6]=(_Float16)c1.z;  bm[7]=(_Float16)c1.w;
            bm[8]=(_Float16)c2.x;  bm[9]=(_Float16)c2.y;
            bm[10]=(_Float16)c2.z; bm[11]=(_Float16)c2.w;
            bm[12]=(_Float16)c3.x; bm[13]=(_Float16)c3.y;
            bm[14]=(_Float16)c3.z; bm[15]=(_Float16)c3.w;

            acc = __builtin_amdgcn_wmma_f32_16x16x32_f16(
                false, a, false, bm, (short)0, acc, false, false);
        }

        // D[m][n] == dot(row m, c) for all n. Lanes 0..15: M = vgpr,
        // lanes 16..31: M = vgpr+8. Sum rows via lanes 0 and 16 only.
        float s = acc[0] + acc[1] + acc[2] + acc[3]
                + acc[4] + acc[5] + acc[6] + acc[7];
        tile_sum = __shfl(s, 0, 32) + __shfl(s, 16, 32);
    }
    if (lane == 0) partials[waveId] = tile_sum;     // exactly one writer
}

// -------------------- Kernel 3: deterministic finalize ----------------------
__global__ __launch_bounds__(256)
void hmm_final(const float* __restrict__ ws,
               const float* __restrict__ partials,
               float* __restrict__ out)
{
    __shared__ float red[256];
    const int tid = threadIdx.x;
    float s = 0.f;
    for (int i = tid; i < N_TILES; i += 256) s += partials[i];  // fixed order
    red[tid] = s; __syncthreads();
    for (int st = 128; st > 0; st >>= 1) {
        if (tid < st) red[tid] += red[tid + st];
        __syncthreads();
    }
    if (tid == 0) {
        const float total_dot = red[0];
        const float S     = ws[WS_S];
        const float kl0   = ws[WS_KL0];
        const float klt   = ws[WS_KLT];
        const float denom = ws[WS_DENOM];
        out[0] = -(S + total_dot / denom);                          // rec_loss
        out[1] = ((float)HMM_B * kl0 + (denom - (float)HMM_B) * klt) / denom;
    }
}

// ------------------------------- launcher -----------------------------------
extern "C" void kernel_launch(void* const* d_in, const int* in_sizes, int n_in,
                              void* d_out, int out_size, void* d_ws, size_t ws_size,
                              hipStream_t stream) {
    (void)in_sizes; (void)n_in; (void)out_size; (void)ws_size;
    const float* x      = (const float*)d_in[0];
    const int*   xlens  = (const int*)  d_in[1];
    const float* trans  = (const float*)d_in[2];
    const float* emis   = (const float*)d_in[3];
    const float* init   = (const float*)d_in[4];
    const float* qlog   = (const float*)d_in[5];
    float* ws  = (float*)d_ws;
    float* out = (float*)d_out;

    hmm_prep<<<1, 512, 0, stream>>>(xlens, trans, emis, init, qlog, ws);
    // 4096 tiles, 8 waves per 256-thread block -> 512 blocks
    hmm_dot<<<N_TILES / 8, 256, 0, stream>>>(x, xlens, ws, ws + WS_PART);
    hmm_final<<<1, 256, 0, stream>>>(ws, ws + WS_PART, out);
}
